// KmeansVectorQuantizer_28887950033231
// MI455X (gfx1250) — matrix operand order repaired and checked by reading, not compiled, not run on hardware
//
#include <hip/hip_runtime.h>
#include <hip/hip_bf16.h>

// ---------------------------------------------------------------------------
// KmeansVectorQuantizer forward for MI455X (gfx1250, wave32, WMMA).
// Bandwidth-bound (~0.7GB traffic @ 23.3TB/s); GEMMs use v_wmma_f32_16x16x32_bf16
// with hi/lo bf16 error compensation (3 WMMAs per 16x16x32 tile -> ~fp32 accuracy).
// bf16 conversion uses native hardware cvt (RNE) instead of bit-twiddling.
// Shapes: B=16 T=4096 D=1024 G=2 K=320 Cg=256 Dc=512, M=B*T=65536.
// ---------------------------------------------------------------------------

#define Mrows 65536
#define Dmod 1024
#define Gg 2
#define Kcode 320
#define Cgrp 256
#define Dc 512

typedef __attribute__((ext_vector_type(16))) __bf16 v16bf;
typedef __attribute__((ext_vector_type(8)))  float  v8f;

struct Frag { v16bf hi, lo; };

__device__ __forceinline__ void split(float f, Frag& fr, int j) {
  __bf16 h = (__bf16)f;                    // native v_cvt (RNE)
  fr.hi[j] = h;
  fr.lo[j] = (__bf16)(f - (float)h);       // residual term
}

// A fragment (16x32, bf16): lane holds row m0+(lane&15); half-wave sel=(lane>>4)
// K pattern: elems 0..7 -> k0+sel*8+j ; elems 8..15 -> k0+16+sel*8+(j-8)
__device__ __forceinline__ Frag load_fragA(const float* base, int ld, int m0, int k0) {
  int lane = threadIdx.x & 31;
  const float* p = base + (size_t)(m0 + (lane & 15)) * ld + k0 + ((lane >> 4) << 3);
  Frag f;
#pragma unroll
  for (int j = 0; j < 8; ++j) split(p[j], f, j);
#pragma unroll
  for (int j = 0; j < 8; ++j) split(p[16 + j], f, 8 + j);
  return f;
}

// B fragment (32x16, bf16), B[k][n] stored as Bmem[n][k] (contiguous k):
// lane holds col n0+(lane&15); elems 0..7 -> k0+sel*8+j ; elems 8..15 -> k0+16+sel*8+(j-8)
__device__ __forceinline__ Frag load_fragB(const float* base, int ld, int n0, int k0) {
  int lane = threadIdx.x & 31;
  const float* p = base + (size_t)(n0 + (lane & 15)) * ld + k0 + ((lane >> 4) << 3);
  Frag f;
#pragma unroll
  for (int j = 0; j < 8; ++j) split(p[j], f, j);
#pragma unroll
  for (int j = 0; j < 8; ++j) split(p[16 + j], f, 8 + j);
  return f;
}

// 3-term bf16 product: a_hi*b_hi + a_hi*b_lo + a_lo*b_hi  (~fp32 fidelity)
__device__ __forceinline__ v8f wmma3(const Frag& a, const Frag& b, v8f c) {
  c = __builtin_amdgcn_wmma_f32_16x16x32_bf16(false, a.lo, false, b.hi, (short)0, c, false, false);
  c = __builtin_amdgcn_wmma_f32_16x16x32_bf16(false, a.hi, false, b.lo, (short)0, c, false, false);
  c = __builtin_amdgcn_wmma_f32_16x16x32_bf16(false, a.hi, false, b.hi, (short)0, c, false, false);
  return c;
}

// --------------------- K1: grouped 1x1 conv GEMM ---------------------------
// Ze[g][row][c] = sum_i Z[row][g*512+i] * conv_w[g][c][i]
__global__ void __launch_bounds__(256)
k_conv(const float* __restrict__ Z, const float* __restrict__ W,
       float* __restrict__ Ze) {
  int wid = threadIdx.x >> 5;                 // 8 waves
  int m0 = blockIdx.x * 128 + wid * 16;
  int g  = blockIdx.y >> 2;
  int n0 = (blockIdx.y & 3) * 64;
  const float* A = Z + g * 512;               // ld = 1024
  const float* B = W + (size_t)g * 256 * 512; // [c][i], ld = 512
  v8f acc[4] = {};
  for (int k0 = 0; k0 < 512; k0 += 32) {
    Frag a = load_fragA(A, Dmod, m0, k0);
#pragma unroll
    for (int t = 0; t < 4; ++t) {
      Frag b = load_fragB(B, 512, n0 + t * 16, k0);
      acc[t] = wmma3(a, b, acc[t]);
    }
  }
  int lane = threadIdx.x & 31;
  int colb = n0 + (lane & 15);
  int rowb = m0 + ((lane >> 4) << 3);
  float* C = Ze + (size_t)g * Mrows * Cgrp;
#pragma unroll
  for (int t = 0; t < 4; ++t)
#pragma unroll
    for (int r = 0; r < 8; ++r)
      C[(size_t)(rowb + r) * Cgrp + colb + t * 16] = acc[t][r];
}

// --------------------- K2: GroupNorm stats (fp64, deterministic) -----------
__global__ void __launch_bounds__(1024)
k_gnstats(const float* __restrict__ Ze, double* __restrict__ stats) {
  int bg = blockIdx.x;                        // b*G+g, 32 blocks
  int b = bg >> 1, g = bg & 1;
  const float* p = Ze + (size_t)g * Mrows * Cgrp + (size_t)b * 4096 * Cgrp;
  double s = 0.0, s2 = 0.0;
  for (int i = threadIdx.x; i < 4096 * Cgrp; i += 1024) {
    double v = (double)p[i];
    s += v; s2 += v * v;
  }
  __shared__ double r0[1024], r1[1024];
  r0[threadIdx.x] = s; r1[threadIdx.x] = s2; __syncthreads();
  for (int off = 512; off > 0; off >>= 1) {
    if ((int)threadIdx.x < off) { r0[threadIdx.x] += r0[threadIdx.x + off];
                                  r1[threadIdx.x] += r1[threadIdx.x + off]; }
    __syncthreads();
  }
  if (threadIdx.x == 0) { stats[bg * 2] = r0[0]; stats[bg * 2 + 1] = r1[0]; }
}

// --------------------- codebook row norms ----------------------------------
__global__ void __launch_bounds__(256)
k_cnorm(const float* __restrict__ codex, float* __restrict__ cnorm) {
  const float* p = codex + (size_t)blockIdx.x * Cgrp;   // 640 blocks
  float v = p[threadIdx.x];
  __shared__ float red[256];
  red[threadIdx.x] = v * v; __syncthreads();
  for (int off = 128; off > 0; off >>= 1) {
    if ((int)threadIdx.x < off) red[threadIdx.x] += red[threadIdx.x + off];
    __syncthreads();
  }
  if (threadIdx.x == 0) cnorm[blockIdx.x] = red[0];
}

// --------------------- K3: distance GEMM + argmin --------------------------
__global__ void __launch_bounds__(256)
k_dist(const float* __restrict__ Ze, const float* __restrict__ codex,
       const float* __restrict__ cnorm, const double* __restrict__ stats,
       const float* __restrict__ gnw, const float* __restrict__ gnb,
       int* __restrict__ idxout) {
  int g   = blockIdx.y;
  int wid = threadIdx.x >> 5;
  int mt  = wid >> 2;                      // m sub-tile 0/1
  int cg  = wid & 3;                       // column group of 80 codes
  int m0  = blockIdx.x * 32 + mt * 16;
  int lane = threadIdx.x & 31;
  int b = (blockIdx.x * 32) >> 12;
  double cnt = 4096.0 * 256.0;
  double mean = stats[(b * 2 + g) * 2] / cnt;
  double var  = stats[(b * 2 + g) * 2 + 1] / cnt - mean * mean;
  float fm = (float)mean;
  float fr = rsqrtf((float)var + 1e-5f);

  const float* A  = Ze + (size_t)g * Mrows * Cgrp;
  const float* Bm = codex + (size_t)g * Kcode * Cgrp;
  v8f acc[5] = {};
  for (int k0 = 0; k0 < 256; k0 += 32) {
    Frag a;
    {
      const float* p = A + (size_t)(m0 + (lane & 15)) * Cgrp + k0 + ((lane >> 4) << 3);
      int cb = k0 + ((lane >> 4) << 3);
#pragma unroll
      for (int j = 0; j < 8; ++j) {
        int c = cb + j;
        split((p[j] - fm) * fr * gnw[g * Cgrp + c] + gnb[g * Cgrp + c], a, j);
      }
#pragma unroll
      for (int j = 0; j < 8; ++j) {
        int c = cb + 16 + j;
        split((p[16 + j] - fm) * fr * gnw[g * Cgrp + c] + gnb[g * Cgrp + c], a, 8 + j);
      }
    }
#pragma unroll
    for (int t = 0; t < 5; ++t) {
      Frag bfr = load_fragB(Bm, Cgrp, cg * 80 + t * 16, k0);
      acc[t] = wmma3(a, bfr, acc[t]);
    }
  }
  // score = ||c||^2 - 2 Zn.c ; argmin with first-index tie-break via u64 key min
  __shared__ unsigned long long cand[32][64];
#pragma unroll
  for (int r = 0; r < 8; ++r) {
    int rowit = r + ((lane >> 4) << 3);
    unsigned long long best = ~0ull;
#pragma unroll
    for (int t = 0; t < 5; ++t) {
      int k = cg * 80 + t * 16 + (lane & 15);
      float score = cnorm[g * Kcode + k] - 2.0f * acc[t][r];
      unsigned u = __builtin_bit_cast(unsigned, score);
      u ^= (u & 0x80000000u) ? 0xFFFFFFFFu : 0x80000000u;
      unsigned long long key = ((unsigned long long)u << 32) | (unsigned)k;
      best = key < best ? key : best;
    }
    cand[mt * 16 + rowit][cg * 16 + (lane & 15)] = best;
  }
  __syncthreads();
  if (threadIdx.x < 32) {
    unsigned long long best = ~0ull;
#pragma unroll 4
    for (int j = 0; j < 64; ++j) {
      unsigned long long k = cand[threadIdx.x][j];
      best = k < best ? k : best;
    }
    int row = blockIdx.x * 32 + threadIdx.x;
    idxout[row * 2 + g] = (int)(best & 0xFFFFFFFFull);
  }
}

// --------------------- one-hot scatter -------------------------------------
__global__ void __launch_bounds__(256)
k_probs(const int* __restrict__ idx, float* __restrict__ probs) {
  int i = blockIdx.x * 256 + threadIdx.x;     // over M*G
  if (i < Mrows * Gg) probs[(size_t)i * Kcode + idx[i]] = 1.0f;
}

// --------------------- K4: loss partials + LayerNorm stats -----------------
__global__ void __launch_bounds__(128)
k_ln(const float* __restrict__ Ze, const float* __restrict__ codex,
     const int* __restrict__ idx, const double* __restrict__ stats,
     const float* __restrict__ gnw, const float* __restrict__ gnb,
     float* __restrict__ lnstats, float* __restrict__ lossPart) {
  int row = blockIdx.x;
  int b = row >> 12;
  int tid = threadIdx.x;                      // 128 threads
  int i0 = idx[row * 2], i1 = idx[row * 2 + 1];
  float fm[2], frv[2];
#pragma unroll
  for (int g = 0; g < 2; ++g) {
    double cnt = 4096.0 * 256.0;
    double mean = stats[(b * 2 + g) * 2] / cnt;
    double var  = stats[(b * 2 + g) * 2 + 1] / cnt - mean * mean;
    fm[g] = (float)mean; frv[g] = rsqrtf((float)var + 1e-5f);
  }
  float s = 0.f, s2 = 0.f, d2 = 0.f;
#pragma unroll
  for (int j = 0; j < 4; ++j) {
    int c = tid + 128 * j;
    int g = c >> 8, cc = c & 255;
    float q  = codex[((size_t)(g * Kcode + (g ? i1 : i0))) * Cgrp + cc];
    float ze = Ze[(size_t)g * Mrows * Cgrp + (size_t)row * Cgrp + cc];
    float zn = (ze - fm[g]) * frv[g] * gnw[c] + gnb[c];
    float d = zn - q;
    d2 += d * d; s += q; s2 += q * q;
  }
  __shared__ float red[3][128];
  red[0][tid] = s; red[1][tid] = s2; red[2][tid] = d2; __syncthreads();
  for (int off = 64; off > 0; off >>= 1) {
    if (tid < off) { red[0][tid] += red[0][tid + off];
                     red[1][tid] += red[1][tid + off];
                     red[2][tid] += red[2][tid + off]; }
    __syncthreads();
  }
  if (tid == 0) {
    float m = red[0][0] * (1.0f / Dc);
    float v = red[1][0] * (1.0f / Dc) - m * m;
    lnstats[row * 2] = m;
    lnstats[row * 2 + 1] = rsqrtf(v + 1e-5f);
    lossPart[row] = red[2][0];
  }
}

// --------------------- K5: post projection GEMM ----------------------------
// A rows regenerated on the fly: gather Zq from codex, apply LayerNorm.
__global__ void __launch_bounds__(256)
k_post(const float* __restrict__ codex, const int* __restrict__ idx,
       const float* __restrict__ lnstats, const float* __restrict__ lnw,
       const float* __restrict__ lnb, const float* __restrict__ Wp,
       const float* __restrict__ bias, float* __restrict__ out) {
  int wid = threadIdx.x >> 5;
  int lane = threadIdx.x & 31;
  int m0 = blockIdx.x * 128 + wid * 16;
  int n0 = blockIdx.y * 64;
  int r  = m0 + (lane & 15);
  float lm = lnstats[r * 2], lr = lnstats[r * 2 + 1];
  int i0 = idx[r * 2], i1 = idx[r * 2 + 1];
  v8f acc[4] = {};
  for (int k0 = 0; k0 < Dc; k0 += 32) {
    Frag a;
    int cb = k0 + ((lane >> 4) << 3);
#pragma unroll
    for (int h = 0; h < 2; ++h) {
#pragma unroll
      for (int j = 0; j < 8; ++j) {
        int c = cb + h * 16 + j;
        int g = c >> 8;
        float q = codex[((size_t)(g * Kcode + (g ? i1 : i0))) * Cgrp + (c & 255)];
        split((q - lm) * lr * lnw[c] + lnb[c], a, h * 8 + j);
      }
    }
#pragma unroll
    for (int t = 0; t < 4; ++t) {
      Frag bf = load_fragB(Wp, Dc, n0 + t * 16, k0);
      acc[t] = wmma3(a, bf, acc[t]);
    }
  }
  int colb = n0 + (lane & 15);
  int rowb = m0 + ((lane >> 4) << 3);
#pragma unroll
  for (int t = 0; t < 4; ++t) {
    float bv = bias[colb + t * 16];
#pragma unroll
    for (int rr = 0; rr < 8; ++rr)
      out[(size_t)(rowb + rr) * Dmod + colb + t * 16] = acc[t][rr] + bv;
  }
}

// --------------------- loss reduction --------------------------------------
__global__ void __launch_bounds__(256)
k_lossred(const float* __restrict__ part, float* __restrict__ outLoss) {
  __shared__ float red[256];
  float s = 0.f;
  for (int i = threadIdx.x; i < Mrows; i += 256) s += part[i];
  red[threadIdx.x] = s; __syncthreads();
  for (int off = 128; off > 0; off >>= 1) {
    if ((int)threadIdx.x < off) red[threadIdx.x] += red[threadIdx.x + off];
    __syncthreads();
  }
  // loss = (BETA + 1) * mean((Zn-Zq)^2) over B*T*G*Cg = 33554432 elements
  if (threadIdx.x == 0) *outLoss = (float)(1.25 * (double)red[0] / 33554432.0);
}

extern "C" void kernel_launch(void* const* d_in, const int* in_sizes, int n_in,
                              void* d_out, int out_size, void* d_ws, size_t ws_size,
                              hipStream_t stream) {
  const float* Z      = (const float*)d_in[0];
  const float* conv_w = (const float*)d_in[1];
  const float* gnw    = (const float*)d_in[2];
  const float* gnb    = (const float*)d_in[3];
  const float* codex  = (const float*)d_in[4];
  const float* lnw    = (const float*)d_in[5];
  const float* lnb    = (const float*)d_in[6];
  const float* postw  = (const float*)d_in[7];
  const float* postb  = (const float*)d_in[8];

  float* out      = (float*)d_out;                       // (B,T,D) = 67108864
  float* outLoss  = out + (size_t)67108864;              // scalar
  float* outProbs = outLoss + 1;                         // (B,T,G,K) = 41943040

  // workspace layout (~135.5 MB)
  char* ws = (char*)d_ws;
  float*  Ze       = (float*)(ws);                       // 134217728 B
  int*    idx      = (int*)  (ws + 134217728);           //    524288 B
  float*  lnstats  = (float*)(ws + 134742016);           //    524288 B
  double* stats    = (double*)(ws + 135266304);          //       512 B
  float*  cnorm    = (float*)(ws + 135266816);           //      2560 B
  float*  lossPart = (float*)(ws + 135269376);           //    262144 B

  hipMemsetAsync(outProbs, 0, (size_t)41943040 * sizeof(float), stream);

  k_cnorm  <<<dim3(Gg * Kcode), 256, 0, stream>>>(codex, cnorm);
  k_conv   <<<dim3(Mrows / 128, 8), 256, 0, stream>>>(Z, conv_w, Ze);
  k_gnstats<<<dim3(32), 1024, 0, stream>>>(Ze, stats);
  k_dist   <<<dim3(Mrows / 32, Gg), 256, 0, stream>>>(Ze, codex, cnorm, stats, gnw, gnb, idx);
  k_probs  <<<dim3((Mrows * Gg + 255) / 256), 256, 0, stream>>>(idx, outProbs);
  k_ln     <<<dim3(Mrows), 128, 0, stream>>>(Ze, codex, idx, stats, gnw, gnb, lnstats, lossPart);
  k_post   <<<dim3(Mrows / 128, Dmod / 64), 256, 0, stream>>>(codex, idx, lnstats, lnw, lnb,
                                                              postw, postb, out);
  k_lossred<<<dim3(1), 256, 0, stream>>>(lossPart, outLoss);
}
